// EncoderLayer_2052994367605
// MI455X (gfx1250) — matrix operand order repaired and checked by reading, not compiled, not run on hardware
//
#include <hip/hip_runtime.h>
#include <cstdint>

#define DEVC __device__ __forceinline__

typedef __bf16 bf16_t;
typedef __attribute__((ext_vector_type(16))) __bf16 v16bf;
typedef __attribute__((ext_vector_type(8)))  __bf16 v8bf;
typedef __attribute__((ext_vector_type(8)))  float  v8f;
typedef __attribute__((ext_vector_type(4)))  float  v4f;
typedef __attribute__((ext_vector_type(4)))  int    v4i;

static constexpr int Bn = 2, Sn = 2048, Dn = 1024, Hn = 16, FFn = 4096, HDn = 64;

#if defined(__has_builtin)
#if __has_builtin(__builtin_amdgcn_global_load_async_to_lds_b128) && \
    __has_builtin(__builtin_amdgcn_s_wait_asynccnt)
#define HAVE_ASYNC_LDS 1
#endif
#endif
#ifndef HAVE_ASYNC_LDS
#define HAVE_ASYNC_LDS 0
#endif

#if HAVE_ASYNC_LDS
typedef __attribute__((address_space(1))) v4i as1_v4i;
typedef __attribute__((address_space(3))) v4i as3_v4i;
// async 16B global -> LDS copy (ASYNCcnt-tracked, bypasses VGPRs)
DEVC void async_copy_b128(const bf16_t* g, bf16_t* l) {
  __builtin_amdgcn_global_load_async_to_lds_b128(
      (as1_v4i*)(uintptr_t)g, (as3_v4i*)(uint32_t)(uintptr_t)l, 0, 0);
}
#endif

DEVC v16bf cat16(v8bf lo, v8bf hi) {
  return __builtin_shufflevector(lo, hi, 0, 1, 2, 3, 4, 5, 6, 7, 8, 9, 10, 11, 12, 13, 14, 15);
}
DEVC v8f vzero8() {
  v8f z = {0.f, 0.f, 0.f, 0.f, 0.f, 0.f, 0.f, 0.f};
  return z;
}
DEVC v8f wmma_bf16(v16bf a, v16bf b, v8f c) {
  return __builtin_amdgcn_wmma_f32_16x16x32_bf16(false, a, false, b, (short)0, c, false, false);
}

// ---------------------------------------------------------------------------
// f32 -> bf16 conversion (4 elements / thread)
// ---------------------------------------------------------------------------
__global__ __launch_bounds__(256) void cvt_f32_to_bf16(const float* __restrict__ src,
                                                       bf16_t* __restrict__ dst, int n) {
  int i = (blockIdx.x * 256 + threadIdx.x) * 4;
  if (i >= n) return;
  v4f f = *(const v4f*)(src + i);
  dst[i + 0] = (bf16_t)f[0];
  dst[i + 1] = (bf16_t)f[1];
  dst[i + 2] = (bf16_t)f[2];
  dst[i + 3] = (bf16_t)f[3];
}

// ---------------------------------------------------------------------------
// General batched bf16 WMMA GEMM: C = A(MxK) * B(KxN) + bias, optional ReLU.
// Block tile 128 x (32*NT); 8 waves in a 4x2 grid, wave tile 32 x (16*NT).
// NT=2 -> block 128x64 (QKV, N=64); NT=4 -> block 128x128 (FFN).
// A batch offset = (z / zdivA) * sA ; B/bias batch offset = (z % zmodB) * stride.
// ---------------------------------------------------------------------------
template <int NT, bool RELU, typename OutT>
__global__ __launch_bounds__(256) void gemm_bf16_kernel(
    const bf16_t* __restrict__ A, long sA, int lda, int zdivA,
    const bf16_t* __restrict__ Bw, long sB, int ldb, int zmodB,
    OutT* __restrict__ C, long sC, int ldc,
    const float* __restrict__ bias, int sBias,
    int M, int N, int K) {
  __shared__ bf16_t ldsA[128][32];        // [m][k]
  __shared__ bf16_t ldsBt[32 * NT][32];   // [n][k] (transposed: B frags k-contiguous)

  const int z = blockIdx.z;
  A += (long)(z / zdivA) * sA;
  Bw += (long)(z % zmodB) * sB;
  const float* bp = bias + (long)(z % zmodB) * sBias;
  C += (long)z * sC;

  const int tid = threadIdx.x;
  const int lane = tid & 31;
  const int wid = tid >> 5;
  const int wrow = wid & 3;   // 4 wave rows
  const int wcol = wid >> 2;  // 2 wave cols
  const int m0 = blockIdx.y * 128;
  const int n0 = blockIdx.x * (32 * NT);
  const int lm = lane & 15;
  const int half = lane >> 4;
  const int abase = half ? 16 : 0;   // A-frag K base (K contiguous per row)
  const int bbase = half ? 16 : 0;   // B-frag K base (K contiguous per column)

  v8f acc[2][NT];
#pragma unroll
  for (int i = 0; i < 2; ++i)
#pragma unroll
    for (int j = 0; j < NT; ++j) acc[i][j] = vzero8();

  const int arow = tid >> 1;
  const int acol = (tid & 1) * 16;
  const int bk = tid >> 3;         // 0..31
  const int bn0 = (tid & 7) * 8;   // 0..56

  for (int k0 = 0; k0 < K; k0 += 32) {
    // stage A tile 128x32 (each thread 32B, coalesced)
    const bf16_t* asrc = A + (long)(m0 + arow) * lda + k0 + acol;
#if HAVE_ASYNC_LDS
    async_copy_b128(asrc, &ldsA[arow][acol]);
    async_copy_b128(asrc + 8, &ldsA[arow][acol + 8]);
#else
    *(v8bf*)&ldsA[arow][acol] = *(const v8bf*)asrc;
    *(v8bf*)&ldsA[arow][acol + 8] = *(const v8bf*)(asrc + 8);
#endif

    // stage B tile 32 x 32*NT transposed into [n][k]
#pragma unroll
    for (int c = 0; c < NT / 2; ++c) {
      const bf16_t* bsrc = Bw + (long)(k0 + bk) * ldb + n0 + bn0 + c * 64;
      v8bf bvv = *(const v8bf*)bsrc;
#pragma unroll
      for (int i = 0; i < 8; ++i) ldsBt[bn0 + c * 64 + i][bk] = bvv[i];
    }

    if (k0 + 32 < K) {
      __builtin_prefetch(A + (long)(m0 + arow) * lda + k0 + 32 + acol, 0, 1);
      __builtin_prefetch(Bw + (long)(k0 + 32 + bk) * ldb + n0 + bn0, 0, 1);
    }
#if HAVE_ASYNC_LDS
    __builtin_amdgcn_s_wait_asynccnt(0);
#endif
    __syncthreads();

    v16bf afr[2], bfr[NT];
#pragma unroll
    for (int mt = 0; mt < 2; ++mt) {
      const bf16_t* p = &ldsA[wrow * 32 + mt * 16 + lm][0];
      afr[mt] = cat16(*(const v8bf*)(p + abase), *(const v8bf*)(p + abase + 8));
    }
#pragma unroll
    for (int nt = 0; nt < NT; ++nt) {
      const bf16_t* p = &ldsBt[wcol * 16 * NT + nt * 16 + lm][0];
      bfr[nt] = cat16(*(const v8bf*)(p + bbase), *(const v8bf*)(p + bbase + 8));
    }
#pragma unroll
    for (int mt = 0; mt < 2; ++mt)
#pragma unroll
      for (int nt = 0; nt < NT; ++nt)
        acc[mt][nt] = wmma_bf16(afr[mt], bfr[nt], acc[mt][nt]);
    __syncthreads();
  }

#pragma unroll
  for (int mt = 0; mt < 2; ++mt)
#pragma unroll
    for (int nt = 0; nt < NT; ++nt) {
      const int col = n0 + wcol * 16 * NT + nt * 16 + lm;
      const float bb = bp[col];
#pragma unroll
      for (int r = 0; r < 8; ++r) {
        const int row = m0 + wrow * 32 + mt * 16 + half * 8 + r;
        float v = acc[mt][nt][r] + bb;
        if (RELU) v = v > 0.f ? v : 0.f;
        C[(long)row * ldc + col] = (OutT)v;
      }
    }
}

// ---------------------------------------------------------------------------
// Flash attention: per (b,h), 64 query rows per block (4 waves x 16 rows).
// Online softmax in f32; the softmax denominator is accumulated by the same
// P*V WMMA via a ones-column appended to V^T (5th accumulator) -- no sum
// reductions needed, only the row-max shuffle reduction.
// ---------------------------------------------------------------------------
__global__ __launch_bounds__(128) void flash_attn_kernel(const bf16_t* __restrict__ Qm,
                                                         const bf16_t* __restrict__ Km,
                                                         const bf16_t* __restrict__ Vm,
                                                         float* __restrict__ O) {
  __shared__ bf16_t ldsK[32][64];      // [key][hd]
  __shared__ bf16_t ldsVt[80][32];     // [hd 0..63 | ones 64..79][key]
  __shared__ bf16_t ldsP[4][16][32];   // per-wave P scratch [m][key]

  const int bh = blockIdx.y;
  const int b = bh >> 4;   // H = 16
  const int h = bh & 15;
  const bf16_t* q = Qm + (long)bh * Sn * HDn;
  const bf16_t* kp = Km + (long)bh * Sn * HDn;
  const bf16_t* vp = Vm + (long)bh * Sn * HDn;

  const int tid = threadIdx.x;
  const int lane = tid & 31;
  const int w = tid >> 5;
  const int lm = lane & 15;
  const int half = lane >> 4;
  const int abase = half ? 16 : 0;
  const int bbase = half ? 16 : 0;
  const int qrow = blockIdx.x * 64 + w * 16 + lm;

  // ones-column region of V^T (static across tiles; covered by first barrier)
#pragma unroll
  for (int i = 0; i < 4; ++i) {
    int idx = tid * 4 + i;  // 0..511
    ldsVt[64 + (idx >> 5)][idx & 31] = (bf16_t)1.0f;
  }

  // Q fragments (A-layout, K = hd 0..31 / 32..63), resident for whole loop
  v16bf aq[2];
#pragma unroll
  for (int f = 0; f < 2; ++f) {
    const bf16_t* p = q + (long)qrow * HDn + f * 32;
    aq[f] = cat16(*(const v8bf*)(p + abase), *(const v8bf*)(p + abase + 8));
  }

  v8f oacc[5];  // [0..3] = O tiles (hd), [4] = softmax denominator
#pragma unroll
  for (int nt = 0; nt < 5; ++nt) oacc[nt] = vzero8();
  float mrow[8];
#pragma unroll
  for (int r = 0; r < 8; ++r) mrow[r] = -1e30f;

  const int krow = tid >> 2;         // 0..31 (key)
  const int kseg = (tid & 3) * 16;   // 0/16/32/48 (hd)

  for (int t = 0; t < Sn / 32; ++t) {
    // ---- stage K tile [key][hd] (coalesced 16B chunks) ----
    const bf16_t* ksrc = kp + (long)(t * 32 + krow) * HDn + kseg;
#if HAVE_ASYNC_LDS
    async_copy_b128(ksrc, &ldsK[krow][kseg]);
    async_copy_b128(ksrc + 8, &ldsK[krow][kseg + 8]);
#else
    *(v8bf*)&ldsK[krow][kseg] = *(const v8bf*)ksrc;
    *(v8bf*)&ldsK[krow][kseg + 8] = *(const v8bf*)(ksrc + 8);
#endif
    // ---- stage V tile transposed: coalesced global load, register transpose ----
    {
      const bf16_t* vsrc = vp + (long)(t * 32 + krow) * HDn + kseg;
      v8bf v0 = *(const v8bf*)vsrc;
      v8bf v1 = *(const v8bf*)(vsrc + 8);
#pragma unroll
      for (int i = 0; i < 8; ++i) ldsVt[kseg + i][krow] = v0[i];
#pragma unroll
      for (int i = 0; i < 8; ++i) ldsVt[kseg + 8 + i][krow] = v1[i];
    }
#if HAVE_ASYNC_LDS
    __builtin_amdgcn_s_wait_asynccnt(0);
#endif
    __syncthreads();

    // ---- S = Q * K^T for 32 keys (two 16-key accumulators) ----
    v8f s0 = vzero8(), s1 = vzero8();
#pragma unroll
    for (int f = 0; f < 2; ++f) {
      const bf16_t* p0 = &ldsK[lm][f * 32 + bbase];
      v16bf bk0 = cat16(*(const v8bf*)p0, *(const v8bf*)(p0 + 8));
      s0 = wmma_bf16(aq[f], bk0, s0);
      const bf16_t* p1 = &ldsK[16 + lm][f * 32 + bbase];
      v16bf bk1 = cat16(*(const v8bf*)p1, *(const v8bf*)(p1 + 8));
      s1 = wmma_bf16(aq[f], bk1, s1);
    }

    // ---- online softmax: row max (16-lane halves), exp, rescale ----
    const float scale = 0.125f;  // 1/sqrt(64)
    float corr[8];
#pragma unroll
    for (int r = 0; r < 8; ++r) {
      float a0 = s0[r] * scale, a1 = s1[r] * scale;
      float rm = fmaxf(a0, a1);
#pragma unroll
      for (int msk = 1; msk < 16; msk <<= 1) rm = fmaxf(rm, __shfl_xor(rm, msk, 32));
      float mnew = fmaxf(mrow[r], rm);
      corr[r] = __expf(mrow[r] - mnew);
      mrow[r] = mnew;
      s0[r] = __expf(a0 - mnew);
      s1[r] = __expf(a1 - mnew);
    }
#pragma unroll
    for (int nt = 0; nt < 5; ++nt)
#pragma unroll
      for (int r = 0; r < 8; ++r) oacc[nt][r] *= corr[r];

    // ---- P (C-layout f32) -> LDS -> A-layout bf16 fragment (same wave) ----
#pragma unroll
    for (int r = 0; r < 8; ++r) {
      int prow = half * 8 + r;
      ldsP[w][prow][lm] = (bf16_t)s0[r];
      ldsP[w][prow][lm + 16] = (bf16_t)s1[r];
    }
    const bf16_t* pp = &ldsP[w][lm][0];
    v16bf ap = cat16(*(const v8bf*)(pp + abase), *(const v8bf*)(pp + abase + 8));

    // ---- O += P * V (tile 4 == ones column -> denominator) ----
#pragma unroll
    for (int nt = 0; nt < 5; ++nt) {
      const bf16_t* pv = &ldsVt[nt * 16 + lm][bbase];
      v16bf bv = cat16(*(const v8bf*)pv, *(const v8bf*)(pv + 8));
      oacc[nt] = wmma_bf16(ap, bv, oacc[nt]);
    }
    __syncthreads();
  }

  // epilogue: O[b, s, h*64 + hd] = acc / denom
  float* outp = O + (long)b * Sn * Dn;
#pragma unroll
  for (int r = 0; r < 8; ++r) {
    const int row = blockIdx.x * 64 + w * 16 + half * 8 + r;
    const float inv = 1.0f / oacc[4][r];
#pragma unroll
    for (int nt = 0; nt < 4; ++nt) {
      const int col = h * HDn + nt * 16 + lm;
      outp[(long)row * Dn + col] = oacc[nt][r] * inv;
    }
  }
}

// ---------------------------------------------------------------------------
// y = LayerNorm(X + R) * gamma + beta ; writes f32 and (optional) bf16 copy.
// One block per row, D = 1024, 256 threads x 4 elements.
// ---------------------------------------------------------------------------
__global__ __launch_bounds__(256) void resid_ln_kernel(const float* __restrict__ X,
                                                       const float* __restrict__ R,
                                                       const float* __restrict__ gamma,
                                                       const float* __restrict__ beta,
                                                       float* __restrict__ Yf,
                                                       bf16_t* __restrict__ Yb) {
  __shared__ float red[8];
  const int row = blockIdx.x;
  const int tid = threadIdx.x;
  const int lane = tid & 31;
  const int wid = tid >> 5;
  const float* x = X + (long)row * Dn;
  const float* r = R + (long)row * Dn;

  float vals[4];
  float s = 0.f;
#pragma unroll
  for (int i = 0; i < 4; ++i) {
    int c = tid + i * 256;
    vals[i] = x[c] + r[c];
    s += vals[i];
  }
#pragma unroll
  for (int m = 1; m < 32; m <<= 1) s += __shfl_xor(s, m, 32);
  if (lane == 0) red[wid] = s;
  __syncthreads();
  float tot = 0.f;
#pragma unroll
  for (int j = 0; j < 8; ++j) tot += red[j];
  const float mu = tot * (1.f / Dn);
  __syncthreads();

  float ss = 0.f;
#pragma unroll
  for (int i = 0; i < 4; ++i) {
    float d = vals[i] - mu;
    ss += d * d;
  }
#pragma unroll
  for (int m = 1; m < 32; m <<= 1) ss += __shfl_xor(ss, m, 32);
  if (lane == 0) red[wid] = ss;
  __syncthreads();
  float tv = 0.f;
#pragma unroll
  for (int j = 0; j < 8; ++j) tv += red[j];
  const float rstd = rsqrtf(tv * (1.f / Dn) + 1e-5f);

#pragma unroll
  for (int i = 0; i < 4; ++i) {
    int c = tid + i * 256;
    float y = (vals[i] - mu) * rstd * gamma[c] + beta[c];
    Yf[(long)row * Dn + c] = y;
    if (Yb) Yb[(long)row * Dn + c] = (bf16_t)y;
  }
}

// ---------------------------------------------------------------------------
// host-side orchestration
// ---------------------------------------------------------------------------
extern "C" void kernel_launch(void* const* d_in, const int* in_sizes, int n_in,
                              void* d_out, int out_size, void* d_ws, size_t ws_size,
                              hipStream_t stream) {
  (void)in_sizes; (void)n_in; (void)out_size; (void)ws_size;
  const float* x   = (const float*)d_in[0];
  const float* Wq  = (const float*)d_in[1];
  const float* bq  = (const float*)d_in[2];
  const float* Wk  = (const float*)d_in[3];
  const float* bk  = (const float*)d_in[4];
  const float* Wv  = (const float*)d_in[5];
  const float* bv  = (const float*)d_in[6];
  const float* g1  = (const float*)d_in[7];
  const float* be1 = (const float*)d_in[8];
  const float* W1  = (const float*)d_in[9];
  const float* b1  = (const float*)d_in[10];
  const float* W2  = (const float*)d_in[11];
  const float* b2  = (const float*)d_in[12];
  const float* g2  = (const float*)d_in[13];
  const float* be2 = (const float*)d_in[14];
  float* out = (float*)d_out;

  char* ws = (char*)d_ws;
  size_t off = 0;
  auto take = [&](size_t bytes) {
    size_t o = off;
    off += (bytes + 255) & ~(size_t)255;
    return o;
  };
  const size_t nXD  = (size_t)Bn * Sn * Dn;        // 4194304
  const size_t nW   = (size_t)Hn * Dn * HDn;       // 1048576
  const size_t nFFW = (size_t)Dn * FFn;            // 4194304
  const size_t nQKV = (size_t)Bn * Hn * Sn * HDn;  // 4194304
  const size_t nH1  = (size_t)Bn * Sn * FFn;       // 16777216

  bf16_t* xb  = (bf16_t*)(ws + take(nXD * 2));
  bf16_t* wqb = (bf16_t*)(ws + take(nW * 2));
  bf16_t* wkb = (bf16_t*)(ws + take(nW * 2));
  bf16_t* wvb = (bf16_t*)(ws + take(nW * 2));
  bf16_t* w1b = (bf16_t*)(ws + take(nFFW * 2));
  bf16_t* w2b = (bf16_t*)(ws + take(nFFW * 2));
  bf16_t* qb  = (bf16_t*)(ws + take(nQKV * 2));
  bf16_t* kb  = (bf16_t*)(ws + take(nQKV * 2));
  bf16_t* vb  = (bf16_t*)(ws + take(nQKV * 2));
  float*  oF  = (float*)(ws + take(nXD * 4));
  float*  y1  = (float*)(ws + take(nXD * 4));
  bf16_t* h1  = (bf16_t*)(ws + take(nH1 * 2));
  bf16_t* y1b = xb;  // alias: xb dead after QKV projections
  float*  ff2 = oF;  // alias: attention output dead after LN1

  // 1) convert everything the GEMMs touch to bf16
  cvt_f32_to_bf16<<<dim3((unsigned)(nXD / 1024)), 256, 0, stream>>>(x, xb, (int)nXD);
  cvt_f32_to_bf16<<<dim3((unsigned)(nW / 1024)), 256, 0, stream>>>(Wq, wqb, (int)nW);
  cvt_f32_to_bf16<<<dim3((unsigned)(nW / 1024)), 256, 0, stream>>>(Wk, wkb, (int)nW);
  cvt_f32_to_bf16<<<dim3((unsigned)(nW / 1024)), 256, 0, stream>>>(Wv, wvb, (int)nW);
  cvt_f32_to_bf16<<<dim3((unsigned)(nFFW / 1024)), 256, 0, stream>>>(W1, w1b, (int)nFFW);
  cvt_f32_to_bf16<<<dim3((unsigned)(nFFW / 1024)), 256, 0, stream>>>(W2, w2b, (int)nFFW);

  // 2) Q/K/V projections: per (b,h): [S,D] x [D,HD] + bias  (block 128x64)
  dim3 gq(HDn / 64, Sn / 128, Bn * Hn);
  gemm_bf16_kernel<2, false, bf16_t><<<gq, 256, 0, stream>>>(
      xb, (long)Sn * Dn, Dn, Hn,
      wqb, (long)Dn * HDn, HDn, Hn,
      qb, (long)Sn * HDn, HDn,
      bq, HDn, Sn, HDn, Dn);
  gemm_bf16_kernel<2, false, bf16_t><<<gq, 256, 0, stream>>>(
      xb, (long)Sn * Dn, Dn, Hn,
      wkb, (long)Dn * HDn, HDn, Hn,
      kb, (long)Sn * HDn, HDn,
      bk, HDn, Sn, HDn, Dn);
  gemm_bf16_kernel<2, false, bf16_t><<<gq, 256, 0, stream>>>(
      xb, (long)Sn * Dn, Dn, Hn,
      wvb, (long)Dn * HDn, HDn, Hn,
      vb, (long)Sn * HDn, HDn,
      bv, HDn, Sn, HDn, Dn);

  // 3) fused attention (softmax never materializes S x S)
  dim3 ga(Sn / 64, Bn * Hn);
  flash_attn_kernel<<<ga, 128, 0, stream>>>(qb, kb, vb, oF);

  // 4) y1 = LN(x + o)  (f32 + bf16 copy for FFN1)
  resid_ln_kernel<<<dim3(Bn * Sn), 256, 0, stream>>>(x, oF, g1, be1, y1, y1b);

  // 5) h1 = ReLU(y1 @ W1 + b1)  (block 128x128)
  dim3 gf1(FFn / 128, (Bn * Sn) / 128, 1);
  gemm_bf16_kernel<4, true, bf16_t><<<gf1, 256, 0, stream>>>(
      y1b, 0, Dn, 1,
      w1b, 0, FFn, 1,
      h1, 0, FFn,
      b1, 0, Bn * Sn, FFn, Dn);

  // 6) ff2 = h1 @ W2 + b2 (f32 out, block 128x128)
  dim3 gf2(Dn / 128, (Bn * Sn) / 128, 1);
  gemm_bf16_kernel<4, false, float><<<gf2, 256, 0, stream>>>(
      h1, 0, FFn, 1,
      w2b, 0, Dn, 1,
      ff2, 0, Dn,
      b2, 0, Bn * Sn, Dn, FFn);

  // 7) out = LN(y1 + ff2)
  resid_ln_kernel<<<dim3(Bn * Sn), 256, 0, stream>>>(y1, ff2, g2, be2, out, nullptr);
}
